// RelationalTransformer_18124761989461
// MI455X (gfx1250) — matrix operand order repaired and checked by reading, not compiled, not run on hardware
//
#include <hip/hip_runtime.h>
#include <hip/hip_bf16.h>

// ---------------- problem constants ----------------
static constexpr int Bc = 8, Sc = 512, Ec = 1024, Hc = 8, DHc = 128,
                     FFc = 4096, Lc = 6, Rc = 33;

// ---------------- WMMA types ----------------
typedef __attribute__((ext_vector_type(16))) __bf16 v16bf;
typedef __attribute__((ext_vector_type(8)))  float  v8f;

union Frag { unsigned u[8]; uint4 q[2]; v16bf v; };

__device__ __forceinline__ unsigned pack2bf(float lo, float hi) {
    union { float f; unsigned u; } a, b;
    a.f = lo; b.f = hi;
    unsigned ua = (a.u + 0x7FFFu + ((a.u >> 16) & 1u)) >> 16;
    unsigned ub = (b.u + 0x7FFFu + ((b.u >> 16) & 1u)) >> 16;
    return (ub << 16) | (ua & 0xFFFFu);
}

// async global->LDS staging helpers (gfx1250 ASYNCcnt path), with sync fallback
#if __has_builtin(__builtin_amdgcn_global_load_async_to_lds_b128)
typedef int rt_v4i __attribute__((ext_vector_type(4)));
typedef __attribute__((address_space(1))) rt_v4i rt_gv4i;   // global (HIP __device__) AS
typedef __attribute__((address_space(3))) rt_v4i rt_lv4i;   // LDS (__shared__) AS
__device__ __forceinline__ void rt_async_copy16(const void* gp, void* lp) {
    __builtin_amdgcn_global_load_async_to_lds_b128((rt_gv4i*)gp, (rt_lv4i*)lp, 0, 0);
}
__device__ __forceinline__ void rt_async_wait() {
#if __has_builtin(__builtin_amdgcn_s_wait_asynccnt)
    __builtin_amdgcn_s_wait_asynccnt(0);
#else
    asm volatile("s_wait_asynccnt 0" ::: "memory");
#endif
}
#else
__device__ __forceinline__ void rt_async_copy16(const void* gp, void* lp) {
    *(uint4*)lp = *(const uint4*)gp;
}
__device__ __forceinline__ void rt_async_wait() {}
#endif

// ---------------- LayerNorm (f32 out, or pair-packed bf16 out) ----------------
__global__ __launch_bounds__(256)
void rt_ln_kernel(const float* __restrict__ x, float* __restrict__ yf,
                  unsigned* __restrict__ yb,
                  const float* __restrict__ g, const float* __restrict__ bt) {
    int row = blockIdx.x, tid = threadIdx.x;
    const float* xr = x + (size_t)row * Ec;
    __shared__ float red[256];
    float s = 0.f;
    for (int i = tid; i < Ec; i += 256) s += xr[i];
    red[tid] = s; __syncthreads();
    for (int off = 128; off > 0; off >>= 1) {
        if (tid < off) red[tid] += red[tid + off];
        __syncthreads();
    }
    float mean = red[0] * (1.0f / Ec);
    __syncthreads();
    float vs = 0.f;
    for (int i = tid; i < Ec; i += 256) { float d = xr[i] - mean; vs += d * d; }
    red[tid] = vs; __syncthreads();
    for (int off = 128; off > 0; off >>= 1) {
        if (tid < off) red[tid] += red[tid + off];
        __syncthreads();
    }
    float rstd = rsqrtf(red[0] * (1.0f / Ec) + 1e-5f);
    if (yb) {   // packed bf16 pairs, row-major [row][Ec/2]
        unsigned* yr = yb + (size_t)row * (Ec / 2);
        for (int i = tid; i < Ec / 2; i += 256) {
            float a0 = (xr[2 * i] - mean) * rstd * g[2 * i] + bt[2 * i];
            float a1 = (xr[2 * i + 1] - mean) * rstd * g[2 * i + 1] + bt[2 * i + 1];
            yr[i] = pack2bf(a0, a1);
        }
    } else {
        float* yr = yf + (size_t)row * Ec;
        for (int i = tid; i < Ec; i += 256)
            yr[i] = (xr[i] - mean) * rstd * g[i] + bt[i];
    }
}

// ---------------- f32 -> pair-packed bf16 (row-major) ----------------
__global__ __launch_bounds__(256)
void rt_pack_kernel(const float* __restrict__ in, unsigned* __restrict__ out, int npairs) {
    int i = blockIdx.x * 256 + threadIdx.x;
    if (i < npairs) {
        const float2 v = *(const float2*)(in + 2 * (size_t)i);
        out[i] = pack2bf(v.x, v.y);
    }
}

// ---------------- weight transpose-pack: W[K,N] f32 -> Wb[N][K/2] packed bf16 ----
__global__ __launch_bounds__(256)
void rt_wpack_kernel(const float* __restrict__ W, unsigned* __restrict__ out,
                     int N, int Khalf) {
    int idx = blockIdx.x * 256 + threadIdx.x;
    if (idx < N * Khalf) {
        int n = idx / Khalf, kp = idx % Khalf;
        out[idx] = pack2bf(W[(size_t)(2 * kp) * N + n], W[(size_t)(2 * kp + 1) * N + n]);
    }
}

// ---------------- bf16-WMMA GEMM, packed operands ----------------
// out[M,N] = op(A[M,K] @ W[K,N] + bias[N]); Ap = [M][K/2] pairs, Wp = [N][K/2] pairs
// flags: 1 = ReLU, 2 = accumulate into out
#define BM 128
#define BN 128
#define BK 32

__global__ __launch_bounds__(256)
void rt_gemm_kernel(const unsigned* __restrict__ Ap, const unsigned* __restrict__ Wp,
                    const float* __restrict__ bias, float* __restrict__ out,
                    int M, int N, int K, int flags) {
    __shared__ __align__(16) unsigned sA[BM][BK / 2];   // 8 KB packed bf16
    __shared__ __align__(16) unsigned sB[BN][BK / 2];   // 8 KB packed bf16
    int tid = threadIdx.x;
    int lane = tid & 31, wave = tid >> 5;
    int wm = wave >> 2, wn = wave & 3;              // 2 x 4 wave grid, wave tile 64x32
    int bm = blockIdx.y * BM, bn = blockIdx.x * BN;
    int m = lane & 15, khalf = lane >> 4;
    int Kh = K / 2;

    v8f acc[4][2] = {};
    int nk = K / BK;
    for (int kb = 0; kb < nk; ++kb) {
        __syncthreads();
        // stage A tile: 128 rows x 16 pairs = 512 x b128 (2 per thread), async cache->LDS
        for (int p = 0; p < 2; ++p) {
            int idx = p * 256 + tid;
            int row = idx >> 2, c4 = (idx & 3) * 4;
            rt_async_copy16(Ap + (size_t)(bm + row) * Kh + kb * 16 + c4,
                            &sA[row][c4]);
        }
        // stage B tile: 128 cols x 16 pairs = 512 x b128
        for (int p = 0; p < 2; ++p) {
            int idx = p * 256 + tid;
            int col = idx >> 2, c4 = (idx & 3) * 4;
            rt_async_copy16(Wp + (size_t)(bn + col) * Kh + kb * 16 + c4,
                            &sB[col][c4]);
        }
        if (kb + 1 < nk)  // gfx1250 global_prefetch_b8 path
            __builtin_prefetch(Ap + (size_t)(bm + (tid >> 1)) * Kh + (kb + 1) * 16, 0, 1);
        rt_async_wait();
        __syncthreads();

        // B fragments: 2 x ds_load_b128 each
        Frag bf[2];
        for (int ni = 0; ni < 2; ++ni) {
            int col = wn * 32 + ni * 16 + m;
            bf[ni].q[0] = *(const uint4*)&sB[col][khalf * 4];
            bf[ni].q[1] = *(const uint4*)&sB[col][8 + khalf * 4];
        }
        // A fragments: 2 x ds_load_b128 each, 8 WMMAs per K-step
        for (int mi = 0; mi < 4; ++mi) {
            Frag a;
            int row = wm * 64 + mi * 16 + m;
            a.q[0] = *(const uint4*)&sA[row][khalf * 4];
            a.q[1] = *(const uint4*)&sA[row][8 + khalf * 4];
            for (int ni = 0; ni < 2; ++ni)
                acc[mi][ni] = __builtin_amdgcn_wmma_f32_16x16x32_bf16(
                    false, a.v, false, bf[ni].v, (short)0, acc[mi][ni], false, false);
        }
    }
    for (int mi = 0; mi < 4; ++mi)
        for (int ni = 0; ni < 2; ++ni)
            for (int r = 0; r < 8; ++r) {
                int rowg = bm + wm * 64 + mi * 16 + khalf * 8 + r;
                int colg = bn + wn * 32 + ni * 16 + m;
                float val = acc[mi][ni][r] + bias[colg];
                if (flags & 1) val = fmaxf(val, 0.f);
                size_t idx = (size_t)rowg * N + colg;
                if (flags & 2) out[idx] += val; else out[idx] = val;
            }
}

// ---------------- fused relation-aware attention ----------------
// one block = (batch b, head h, 16 query rows); 8 waves, 256 threads
__global__ __launch_bounds__(256)
void rt_attn_kernel(const float* __restrict__ Q, const float* __restrict__ Kb,
                    const float* __restrict__ Vb, const int* __restrict__ rel,
                    const int* __restrict__ lengths, const float* __restrict__ relk,
                    const float* __restrict__ relv, float* __restrict__ out) {
    const int tiles = Sc / 16;                       // 32
    int bid = blockIdx.x;
    int b    = bid / (Hc * tiles);
    int rem  = bid % (Hc * tiles);
    int head = rem / tiles;
    int it   = rem % tiles;
    int row0 = it * 16;
    int len  = lengths[b];
    int hc   = head * DHc;

    int tid = threadIdx.x, lane = tid & 31, wave = tid >> 5;
    int m = lane & 15, khalf = lane >> 4;

    __shared__ float sS[16][Sc];        // scores -> attn (32 KB)
    __shared__ float sQr[16][Rc];
    __shared__ float sArel[16][Rc];
    __shared__ float sRed[16][16];
    __shared__ float sMax[16], sSum[16];

    // qr[i][r] = q_i . rel_k[r]
    for (int t = tid; t < 16 * Rc; t += 256) {
        int i = t / Rc, r = t % Rc;
        const float* qp = Q + ((size_t)(b * Sc + row0 + i)) * Ec + hc;
        const float* rp = relk + r * DHc;
        float a = 0.f;
        for (int d = 0; d < DHc; ++d) a += qp[d] * rp[d];
        sQr[i][r] = a;
    }
    __syncthreads();

    // per-wave Q fragments (16 rows x 128 dh, 4 K-chunks of 32)
    Frag aq[4];
    for (int c = 0; c < 4; ++c)
        for (int v = 0; v < 8; ++v) {
            int g = v >> 2, w = v & 3;
            int kp = g * 8 + khalf * 4 + w;
            const float2 qv = *(const float2*)(Q + ((size_t)(b * Sc + row0 + m)) * Ec + hc + c * 32 + 2 * kp);
            aq[c].u[v] = pack2bf(qv.x, qv.y);
        }
    const float scale = 0.088388347648318447f;  // 1/sqrt(128)

    // phase 1: scores = (q.k + qr[rel]) * scale, masked
    for (int jt = wave; jt < tiles; jt += 8) {
        v8f acc = {};
        int key = jt * 16 + m;                 // B-frag column -> key index
        for (int c = 0; c < 4; ++c) {
            Frag bk;
            for (int v = 0; v < 8; ++v) {
                int dh0 = c * 32 + (v >> 2) * 16 + khalf * 8 + 2 * (v & 3);
                const float2 kv = *(const float2*)(Kb + ((size_t)(b * Sc + key)) * Ec + hc + dh0);
                bk.u[v] = pack2bf(kv.x, kv.y);
            }
            acc = __builtin_amdgcn_wmma_f32_16x16x32_bf16(
                false, aq[c].v, false, bk.v, (short)0, acc, false, false);
        }
        int j = jt * 16 + m;
        bool kvalid = j < len;
        const int* relb = rel + (size_t)b * Sc * Sc;
        for (int r = 0; r < 8; ++r) {
            int mm = khalf * 8 + r;
            int ri = relb[(size_t)(row0 + mm) * Sc + j];
            float s = (acc[r] + sQr[mm][ri]) * scale;
            bool valid = kvalid && ((row0 + mm) < len);
            sS[mm][j] = valid ? s : -1e9f;
        }
    }
    __syncthreads();

    // phase 2: softmax over each row of sS
    int srow = tid >> 4, slot = tid & 15;
    float mx = -3.4e38f;
    for (int j = slot; j < Sc; j += 16) mx = fmaxf(mx, sS[srow][j]);
    sRed[srow][slot] = mx; __syncthreads();
    if (slot == 0) {
        float v2 = sRed[srow][0];
        for (int t2 = 1; t2 < 16; ++t2) v2 = fmaxf(v2, sRed[srow][t2]);
        sMax[srow] = v2;
    }
    __syncthreads();
    float rm = sMax[srow], sm = 0.f;
    for (int j = slot; j < Sc; j += 16) {
        float e = __expf(sS[srow][j] - rm);
        sS[srow][j] = e; sm += e;
    }
    sRed[srow][slot] = sm; __syncthreads();
    if (slot == 0) {
        float v2 = 0.f;
        for (int t2 = 0; t2 < 16; ++t2) v2 += sRed[srow][t2];
        sSum[srow] = 1.f / v2;
    }
    __syncthreads();
    float rs = sSum[srow];
    for (int j = slot; j < Sc; j += 16) sS[srow][j] *= rs;

    // phase 3: a_rel[i][r] = sum_j attn[i][j] * [rel==r]
    for (int t = tid; t < 16 * Rc; t += 256) ((float*)sArel)[t] = 0.f;
    __syncthreads();
    for (int j = slot; j < Sc; j += 16) {
        int ri = rel[((size_t)b * Sc + row0 + srow) * Sc + j];
        atomicAdd(&sArel[srow][ri], sS[srow][j]);
    }
    __syncthreads();

    // phase 4: out = attn @ V + a_rel @ rel_v ; wave w owns dh cols [w*16, w*16+16)
    v8f acc = {};
    int d = wave * 16 + m;
    for (int ck = 0; ck < Sc / 32; ++ck) {
        Frag aa, bv;
        for (int v = 0; v < 8; ++v) {
            int g = v >> 2, w = v & 3;
            int kp = g * 8 + khalf * 4 + w;
            int j0 = ck * 32 + 2 * kp;
            aa.u[v] = pack2bf(sS[m][j0], sS[m][j0 + 1]);
        }
        for (int v = 0; v < 8; ++v) {
            int key0 = ck * 32 + (v >> 2) * 16 + khalf * 8 + 2 * (v & 3);
            float f0 = Vb[((size_t)(b * Sc + key0)) * Ec + hc + d];
            float f1 = Vb[((size_t)(b * Sc + key0 + 1)) * Ec + hc + d];
            bv.u[v] = pack2bf(f0, f1);
        }
        acc = __builtin_amdgcn_wmma_f32_16x16x32_bf16(
            false, aa.v, false, bv.v, (short)0, acc, false, false);
    }
    for (int r = 0; r < 8; ++r) {
        int mm = khalf * 8 + r;
        float bias = 0.f;
        for (int rr = 0; rr < Rc; ++rr) bias += sArel[mm][rr] * relv[rr * DHc + d];
        out[((size_t)(b * Sc + row0 + mm)) * Ec + hc + d] = acc[r] + bias;
    }
}

// ---------------- host orchestration ----------------
extern "C" void kernel_launch(void* const* d_in, const int* in_sizes, int n_in,
                              void* d_out, int out_size, void* d_ws, size_t ws_size,
                              hipStream_t stream) {
    (void)in_sizes; (void)n_in; (void)out_size; (void)ws_size;
    const float* enc   = (const float*)d_in[0];
    const int*   rels  = (const int*)  d_in[1];
    const int*   lens  = (const int*)  d_in[2];
    const float* Wq    = (const float*)d_in[3];
    const float* bq    = (const float*)d_in[4];
    const float* Wk    = (const float*)d_in[5];
    const float* bk    = (const float*)d_in[6];
    const float* Wv    = (const float*)d_in[7];
    const float* bv    = (const float*)d_in[8];
    const float* Wo    = (const float*)d_in[9];
    const float* bo    = (const float*)d_in[10];
    const float* rel_k = (const float*)d_in[11];
    const float* rel_v = (const float*)d_in[12];
    const float* W1    = (const float*)d_in[13];
    const float* b1    = (const float*)d_in[14];
    const float* W2    = (const float*)d_in[15];
    const float* b2    = (const float*)d_in[16];
    const float* ln1g  = (const float*)d_in[17];
    const float* ln1b  = (const float*)d_in[18];
    const float* ln2g  = (const float*)d_in[19];
    const float* ln2b  = (const float*)d_in[20];
    const float* lnfg  = (const float*)d_in[21];
    const float* lnfb  = (const float*)d_in[22];

    const size_t NROW = (size_t)Bc * Sc;          // 4096
    char* base = (char*)d_ws;
    const size_t MB = 1024 * 1024;
    float*    x    = (float*)   (base +   0 * MB);   // 16 MB
    float*    q    = (float*)   (base +  16 * MB);   // 16 MB (ff reuses q..ao)
    float*    k    = (float*)   (base +  32 * MB);   // 16 MB (ff reuses q..ao)
    float*    v    = (float*)   (base +  48 * MB);   // 16 MB (ff reuses q..ao)
    float*    ao   = (float*)   (base +  64 * MB);   // 16 MB (ff reuses q..ao)
    unsigned* hb   = (unsigned*)(base +  80 * MB);   // 8 MB packed bf16 LN output
    unsigned* aob  = (unsigned*)(base +  88 * MB);   // 8 MB packed bf16 attn output
    unsigned* ffb  = (unsigned*)(base +  96 * MB);   // 32 MB packed bf16 FFN mid
    unsigned* Wqb  = (unsigned*)(base + 128 * MB);   // 2 MB each
    unsigned* Wkb  = (unsigned*)(base + 130 * MB);
    unsigned* Wvb  = (unsigned*)(base + 132 * MB);
    unsigned* Wob  = (unsigned*)(base + 134 * MB);
    unsigned* W1b  = (unsigned*)(base + 136 * MB);   // 8 MB
    unsigned* W2b  = (unsigned*)(base + 144 * MB);   // 8 MB
    float*    ff   = q;

    (void)hipMemcpyAsync(x, enc, NROW * Ec * sizeof(float),
                         hipMemcpyDeviceToDevice, stream);

    dim3 gProj(Ec / BN, NROW / BM);               // 8 x 32
    dim3 gF1(FFc / BN, NROW / BM);                // 32 x 32
    dim3 gF2(Ec / BN, NROW / BM);
    const int wpProj = (Ec * (Ec / 2)) / 256;     // 2048 blocks
    const int wpFF   = (FFc * (Ec / 2)) / 256;    // 8192 blocks

    for (int l = 0; l < Lc; ++l) {
        const size_t wofs = (size_t)l * Ec * Ec;
        // per-layer weight transpose-pack to bf16 pairs
        rt_wpack_kernel<<<wpProj, 256, 0, stream>>>(Wq + wofs, Wqb, Ec, Ec / 2);
        rt_wpack_kernel<<<wpProj, 256, 0, stream>>>(Wk + wofs, Wkb, Ec, Ec / 2);
        rt_wpack_kernel<<<wpProj, 256, 0, stream>>>(Wv + wofs, Wvb, Ec, Ec / 2);
        rt_wpack_kernel<<<wpProj, 256, 0, stream>>>(Wo + wofs, Wob, Ec, Ec / 2);
        rt_wpack_kernel<<<wpFF, 256, 0, stream>>>(W1 + (size_t)l * Ec * FFc, W1b, FFc, Ec / 2);
        rt_wpack_kernel<<<wpFF, 256, 0, stream>>>(W2 + (size_t)l * FFc * Ec, W2b, Ec, FFc / 2);

        rt_ln_kernel<<<(int)NROW, 256, 0, stream>>>(x, nullptr, hb,
                                                    ln1g + l * Ec, ln1b + l * Ec);
        rt_gemm_kernel<<<gProj, 256, 0, stream>>>(hb, Wqb, bq + l * Ec, q,
                                                  (int)NROW, Ec, Ec, 0);
        rt_gemm_kernel<<<gProj, 256, 0, stream>>>(hb, Wkb, bk + l * Ec, k,
                                                  (int)NROW, Ec, Ec, 0);
        rt_gemm_kernel<<<gProj, 256, 0, stream>>>(hb, Wvb, bv + l * Ec, v,
                                                  (int)NROW, Ec, Ec, 0);
        rt_attn_kernel<<<Bc * Hc * (Sc / 16), 256, 0, stream>>>(
            q, k, v, rels, lens,
            rel_k + (size_t)l * Rc * DHc, rel_v + (size_t)l * Rc * DHc, ao);
        rt_pack_kernel<<<(int)(NROW * Ec / 2 / 256), 256, 0, stream>>>(
            ao, aob, (int)(NROW * Ec / 2));
        rt_gemm_kernel<<<gProj, 256, 0, stream>>>(aob, Wob, bo + l * Ec, x,
                                                  (int)NROW, Ec, Ec, 2);   // += residual
        rt_ln_kernel<<<(int)NROW, 256, 0, stream>>>(x, nullptr, hb,
                                                    ln2g + l * Ec, ln2b + l * Ec);
        rt_gemm_kernel<<<gF1, 256, 0, stream>>>(hb, W1b, b1 + (size_t)l * FFc, ff,
                                                (int)NROW, FFc, Ec, 1);     // ReLU
        rt_pack_kernel<<<(int)(NROW * FFc / 2 / 256), 256, 0, stream>>>(
            ff, ffb, (int)(NROW * FFc / 2));
        rt_gemm_kernel<<<gF2, 256, 0, stream>>>(ffb, W2b, b2 + l * Ec, x,
                                                (int)NROW, Ec, FFc, 2);     // += residual
    }
    rt_ln_kernel<<<(int)NROW, 256, 0, stream>>>(x, (float*)d_out, nullptr, lnfg, lnfb);
}
